// BaselineRNN_32040456028177
// MI455X (gfx1250) — matrix operand order repaired and checked by reading, not compiled, not run on hardware
//
#include <hip/hip_runtime.h>
#include <math.h>

// ---------------------------------------------------------------------------
// Two-layer SimpleRNN + dense head for MI455X (gfx1250, wave32, WMMA).
//
// Kernel 1: xp1 = x @ Wx1 + b1   (big parallel GEMM, f16 WMMA, f32 accum)
//           stored in d_ws pre-swizzled in WMMA C/D-fragment layout (64 MB).
// Kernel 2: fused sequential scan over T for both RNN layers + dense head,
//           using native V_TANH_F32 on the recurrence critical path.
// ---------------------------------------------------------------------------

typedef __attribute__((ext_vector_type(16))) _Float16 v16h;
typedef __attribute__((ext_vector_type(8)))  float    v8f;

#define B_   1024
#define T_   512
#define F_   64
#define H1_  32
#define H2_  16
#define D1_  16
#define D2_  8

// A-fragment (16-bit, 16x32) K index for element j of v16h, given lane half.
// ISA 7.12.2: lanes 0-15: VGPR0-3 -> K=0..7, VGPR4-7 -> K=16..23; lanes 16-31: +8.
__device__ __forceinline__ int kA(int j, int half) {
  return (j & 7) + ((j & 8) << 1) + (half << 3);
}
// B-fragment (Kx16) K index: same K order as the A-fragment (j<8: K=8*half+j,
// j>=8: K=16+8*half+(j-8)).
__device__ __forceinline__ int kB(int j, int half) {
  return (j & 7) + ((j & 8) << 1) + (half << 3);
}

__device__ __forceinline__ v8f wmma_f16(v16h a, v16h b, v8f c) {
  return __builtin_amdgcn_wmma_f32_16x16x32_f16(false, a, false, b,
                                                (short)0, c, false, false);
}

// Native CDNA5 V_TANH_F32 when the toolchain exposes it; libm fallback else.
__device__ __forceinline__ float fast_tanh(float x) {
#if __has_builtin(__builtin_amdgcn_tanhf)
  return __builtin_amdgcn_tanhf(x);
#elif __has_builtin(__builtin_amdgcn_tanh_f32)
  return __builtin_amdgcn_tanh_f32(x);
#else
  return tanhf(x);
#endif
}

// xp fragment address: [t][btile][ntile][lane][8] floats
__device__ __forceinline__ size_t xp_off(int t, int btile, int nt, int lane) {
  return ((((size_t)t * (B_ / 16) + btile) * 2 + nt) * 32 + (size_t)lane) * 8;
}

// ---------------------------------------------------------------------------
// Kernel 1: input projection GEMM.  4096 waves; each wave owns one 16-row
// batch tile and 8 consecutive timesteps (amortizes weight-fragment build).
// ---------------------------------------------------------------------------
__global__ __launch_bounds__(256) void rnn_xproj_kernel(
    const float* __restrict__ x, const float* __restrict__ Wx1,
    const float* __restrict__ b1, float* __restrict__ xp) {
  const int lane   = threadIdx.x & 31;
  const int wave   = threadIdx.x >> 5;
  const int w      = blockIdx.x * 8 + wave;   // 0..4095
  const int btile  = w & 63;                  // 64 tiles of 16 batch rows
  const int tchunk = w >> 6;                  // 0..63, 8 timesteps each
  const int half   = lane >> 4;
  const int m      = lane & 15;

  // Build Wx1 B-fragments: 2 K-chunks (K=0..31, 32..63) x 2 N-tiles (cols 0-15,16-31).
  v16h wf[2][2];
#pragma unroll
  for (int kc = 0; kc < 2; ++kc) {
#pragma unroll
    for (int nt = 0; nt < 2; ++nt) {
      v16h f;
#pragma unroll
      for (int j = 0; j < 16; ++j) {
        int k = kc * 32 + kB(j, half);
        f[j] = (_Float16)Wx1[k * H1_ + nt * 16 + m];
      }
      wf[kc][nt] = f;
    }
  }
  const float bias0 = b1[m];
  const float bias1 = b1[16 + m];

  const float* xrow = x + (size_t)(btile * 16 + m) * T_ * F_;

  for (int ti = 0; ti < 8; ++ti) {
    const int t = tchunk * 8 + ti;
    const float* xr = xrow + (size_t)t * F_;

    // A-fragments for K-chunks: per lane, two contiguous 8-float segments each.
    v16h a[2];
#pragma unroll
    for (int kc = 0; kc < 2; ++kc) {
      const float* p0 = xr + kc * 32 + half * 8;       // K = kc*32 + 8*half + 0..7   (j=0..7)
      const float* p1 = p0 + 16;                       // K = +16                     (j=8..15)
      float4 f0 = *(const float4*)(p0);
      float4 f1 = *(const float4*)(p0 + 4);
      float4 f2 = *(const float4*)(p1);
      float4 f3 = *(const float4*)(p1 + 4);
      v16h av;
      av[0] = (_Float16)f0.x;  av[1] = (_Float16)f0.y;
      av[2] = (_Float16)f0.z;  av[3] = (_Float16)f0.w;
      av[4] = (_Float16)f1.x;  av[5] = (_Float16)f1.y;
      av[6] = (_Float16)f1.z;  av[7] = (_Float16)f1.w;
      av[8]  = (_Float16)f2.x; av[9]  = (_Float16)f2.y;
      av[10] = (_Float16)f2.z; av[11] = (_Float16)f2.w;
      av[12] = (_Float16)f3.x; av[13] = (_Float16)f3.y;
      av[14] = (_Float16)f3.z; av[15] = (_Float16)f3.w;
      a[kc] = av;
    }

    v8f c0, c1;
#pragma unroll
    for (int r = 0; r < 8; ++r) { c0[r] = bias0; c1[r] = bias1; }
    c0 = wmma_f16(a[0], wf[0][0], c0);
    c0 = wmma_f16(a[1], wf[1][0], c0);
    c1 = wmma_f16(a[0], wf[0][1], c1);
    c1 = wmma_f16(a[1], wf[1][1], c1);

    *(v8f*)(xp + xp_off(t, btile, 0, lane)) = c0;
    *(v8f*)(xp + xp_off(t, btile, 1, lane)) = c1;
  }
}

// ---------------------------------------------------------------------------
// Kernel 2: sequential scan, one wave (32 threads) per 16-row batch tile.
// ---------------------------------------------------------------------------
__global__ __launch_bounds__(32) void rnn_scan_kernel(
    const float* __restrict__ xp,
    const float* __restrict__ Wh1, const float* __restrict__ Wx2,
    const float* __restrict__ Wh2, const float* __restrict__ b2,
    const float* __restrict__ W3,  const float* __restrict__ b3,
    const float* __restrict__ W4,  const float* __restrict__ b4,
    const float* __restrict__ Wo,  const float* __restrict__ bo,
    float* __restrict__ out) {
  __shared__ float h1s[16][H1_ + 1];   // h1 tile, row-major, padded
  __shared__ float h2s[16][H2_ + 1];   // h2 tile

  const int lane  = threadIdx.x & 31;
  const int half  = lane >> 4;
  const int m     = lane & 15;
  const int btile = blockIdx.x;        // 64 tiles

  // Recurrent / layer-2 weight B-fragments (built once).
  v16h wh1f[2];
#pragma unroll
  for (int nt = 0; nt < 2; ++nt) {
    v16h f;
#pragma unroll
    for (int j = 0; j < 16; ++j) f[j] = (_Float16)Wh1[kB(j, half) * H1_ + nt * 16 + m];
    wh1f[nt] = f;
  }
  v16h wx2f, wh2f;
#pragma unroll
  for (int j = 0; j < 16; ++j) {
    int k = kB(j, half);
    wx2f[j] = (_Float16)Wx2[k * H2_ + m];
    wh2f[j] = (k < H2_) ? (_Float16)Wh2[k * H2_ + m] : (_Float16)0.0f;  // K padded 16->32
  }
  const float bias2 = b2[m];

  // Zero initial hidden-state A-fragments.
  v16h a1, a2;
#pragma unroll
  for (int j = 0; j < 16; ++j) { a1[j] = (_Float16)0.0f; a2[j] = (_Float16)0.0f; }

  // Preload xp1 fragments for t=0; double-buffer across the loop.
  v8f cur0 = *(const v8f*)(xp + xp_off(0, btile, 0, lane));
  v8f cur1 = *(const v8f*)(xp + xp_off(0, btile, 1, lane));

  for (int t = 0; t < T_; ++t) {
    v8f nxt0, nxt1;
#pragma unroll
    for (int r = 0; r < 8; ++r) { nxt0[r] = 0.0f; nxt1[r] = 0.0f; }
    if (t + 1 < T_) {                       // uniform branch: EXEC stays all-1
      nxt0 = *(const v8f*)(xp + xp_off(t + 1, btile, 0, lane));
      nxt1 = *(const v8f*)(xp + xp_off(t + 1, btile, 1, lane));
    }
    if (t + 6 < T_) {                       // deep prefetch into cache
      __builtin_prefetch(xp + xp_off(t + 6, btile, 0, lane), 0, 3);
      __builtin_prefetch(xp + xp_off(t + 6, btile, 1, lane), 0, 3);
    }

    // h1 = tanh(xp1_t + h1 @ Wh1)
    v8f c0 = wmma_f16(a1, wh1f[0], cur0);
    v8f c1 = wmma_f16(a1, wh1f[1], cur1);
#pragma unroll
    for (int r = 0; r < 8; ++r) {
      h1s[r + 8 * half][m]      = fast_tanh(c0[r]);   // D layout: N=m, M=r+8*half
      h1s[r + 8 * half][16 + m] = fast_tanh(c1[r]);
    }
    __syncthreads();
    // Re-pack h1 into A-fragment layout.
#pragma unroll
    for (int j = 0; j < 16; ++j) a1[j] = (_Float16)h1s[m][kA(j, half)];

    // h2 = tanh(h1 @ Wx2 + b2 + h2 @ Wh2)
    v8f c2;
#pragma unroll
    for (int r = 0; r < 8; ++r) c2[r] = bias2;
    c2 = wmma_f16(a1, wx2f, c2);
    c2 = wmma_f16(a2, wh2f, c2);
    __syncthreads();                        // previous h2s reads done before overwrite
#pragma unroll
    for (int r = 0; r < 8; ++r) h2s[r + 8 * half][m] = fast_tanh(c2[r]);
    __syncthreads();
#pragma unroll
    for (int j = 0; j < 16; ++j) {
      int k = kA(j, half);
      a2[j] = (k < H2_) ? (_Float16)h2s[m][k] : (_Float16)0.0f;
    }

    cur0 = nxt0;
    cur1 = nxt1;
  }

  // Dense head on final h2 (scalar f32; tiny). Lanes 0-15, one batch row each.
  if (lane < 16) {
    const int row = lane;
    float y1[D1_];
#pragma unroll
    for (int d = 0; d < D1_; ++d) y1[d] = b3[d];
#pragma unroll
    for (int k = 0; k < H2_; ++k) {
      const float v = h2s[row][k];
#pragma unroll
      for (int d = 0; d < D1_; ++d) y1[d] = fmaf(v, W3[k * D1_ + d], y1[d]);
    }
#pragma unroll
    for (int d = 0; d < D1_; ++d) y1[d] = fmaxf(y1[d], 0.0f);

    float y2[D2_];
#pragma unroll
    for (int e = 0; e < D2_; ++e) y2[e] = b4[e];
#pragma unroll
    for (int d = 0; d < D1_; ++d)
#pragma unroll
      for (int e = 0; e < D2_; ++e) y2[e] = fmaf(y1[d], W4[d * D2_ + e], y2[e]);

    float o = bo[0];
#pragma unroll
    for (int e = 0; e < D2_; ++e) o = fmaf(y2[e], Wo[e], o);
    out[btile * 16 + row] = o;
  }
}

// ---------------------------------------------------------------------------
extern "C" void kernel_launch(void* const* d_in, const int* in_sizes, int n_in,
                              void* d_out, int out_size, void* d_ws, size_t ws_size,
                              hipStream_t stream) {
  (void)in_sizes; (void)n_in; (void)out_size; (void)ws_size;

  const float* x   = (const float*)d_in[0];
  const float* Wx1 = (const float*)d_in[1];
  const float* Wh1 = (const float*)d_in[2];
  const float* b1  = (const float*)d_in[3];
  const float* Wx2 = (const float*)d_in[4];
  const float* Wh2 = (const float*)d_in[5];
  const float* b2  = (const float*)d_in[6];
  const float* W3  = (const float*)d_in[7];
  const float* b3  = (const float*)d_in[8];
  const float* W4  = (const float*)d_in[9];
  const float* b4  = (const float*)d_in[10];
  const float* Wo  = (const float*)d_in[11];
  const float* bo  = (const float*)d_in[12];
  float* out = (float*)d_out;
  float* xp  = (float*)d_ws;   // needs T_*B_*H1_*4 = 64 MB of workspace

  // Kernel 1: 4096 waves = 512 blocks x 8 waves.
  rnn_xproj_kernel<<<dim3(512), dim3(256), 0, stream>>>(x, Wx1, b1, xp);
  // Kernel 2: one wave per 16-row batch tile.
  rnn_scan_kernel<<<dim3(B_ / 16), dim3(32), 0, stream>>>(
      xp, Wh1, Wx2, Wh2, b2, W3, b3, W4, b4, Wo, bo, out);
}
